// TransformerXL_17867063951633
// MI455X (gfx1250) — compile-verified, matrix operand order checked by reading, not run on hardware
//
#include <hip/hip_runtime.h>
#include <math.h>

// ---------------------------------------------------------------- constants
constexpr int kL  = 4;
constexpr int kD  = 1024;
constexpr int kH  = 16;
constexpr int kDH = 64;
constexpr int kFF = 4096;
constexpr int kV  = 8192;
constexpr int kB  = 4;
constexpr int kQ  = 512;
constexpr int kM  = 512;
constexpr int kE  = 512;
constexpr int kKLEN = kM + kQ;   // 1024
constexpr int kBQ   = kB * kQ;   // 2048

typedef _Float16 h16;
typedef __attribute__((ext_vector_type(16))) _Float16 v16h;
typedef __attribute__((ext_vector_type(8)))  float    v8f;

// ---------------------------------------------------------------- WMMA helpers
// CDNA5 v_wmma_f32_16x16x32_f16 fragment layouts (ISA 7.12.2, wave32):
//  A (16x32 f16): lane = M row (lane&15), lane half selects K group;
//    VGPR i (pair of f16) holds K = frag_k0(i, half), frag_k0+1.
//  B (32x16 f16): symmetric, lane = N column, same K packing.
//  C/D (16x16 f32, 8 VGPRs): lane = N column (lane&15), VGPR i -> M = i + 8*(lane>>4).
__device__ __forceinline__ int frag_k0(int i, int half) {
  return (i < 4) ? (8 * half + 2 * i) : (16 + 8 * half + 2 * (i - 4));
}

__device__ __forceinline__ v8f zero8() {
  v8f z = {0.f, 0.f, 0.f, 0.f, 0.f, 0.f, 0.f, 0.f};
  return z;
}

__device__ __forceinline__ v8f wmma32(v16h a, v16h b, v8f c) {
  return __builtin_amdgcn_wmma_f32_16x16x32_f16(
      /*neg_a=*/false, a, /*neg_b=*/false, b,
      /*c_mod=*/(short)0, c, /*reuse_a=*/false, /*reuse_b=*/false);
}

// Row-major f16 fragment load: p = tile origin, `stride` = elements between rows.
// Serves both A (lane = M row) and B-from-transposed (lane = N col).
// Contiguous 8+8 f16 per lane -> compiler emits 2x global_load_b128.
__device__ __forceinline__ v16h ld_frag(const h16* __restrict__ p, int stride) {
  int lane = threadIdx.x & 31;
  int half = lane >> 4;
  const h16* b = p + (size_t)(lane & 15) * stride;
  v16h f;
#pragma unroll
  for (int i = 0; i < 8; ++i) {
    int k0 = frag_k0(i, half);
    f[2 * i]     = b[k0];
    f[2 * i + 1] = b[k0 + 1];
  }
  return f;
}

// Per-lane prefetch of this fragment's row, `ahead` f16 elements down the K dim.
// Lowers to global_prefetch_b8 (gfx1250).
__device__ __forceinline__ void pf_frag(const h16* __restrict__ p, int stride, int ahead) {
  int lane = threadIdx.x & 31;
  __builtin_prefetch(p + (size_t)(lane & 15) * stride + ahead, 0, 3);
}

// B fragment where the K dimension is strided in memory (lane = N col contiguous).
__device__ __forceinline__ v16h ld_fragB_strided(const h16* __restrict__ p, int kstride) {
  int lane = threadIdx.x & 31;
  int half = lane >> 4;
  const h16* b = p + (lane & 15);
  v16h f;
#pragma unroll
  for (int i = 0; i < 8; ++i) {
    int k0 = frag_k0(i, half);
    f[2 * i]     = b[(size_t)k0 * kstride];
    f[2 * i + 1] = b[(size_t)(k0 + 1) * kstride];
  }
  return f;
}

// A fragment sourced from an f32 LDS buffer (probabilities), converted to f16.
__device__ __forceinline__ v16h ld_fragA_f32(const float* __restrict__ p, int stride) {
  int lane = threadIdx.x & 31;
  int half = lane >> 4;
  const float* b = p + (size_t)(lane & 15) * stride;
  v16h f;
#pragma unroll
  for (int i = 0; i < 8; ++i) {
    int k0 = frag_k0(i, half);
    f[2 * i]     = (h16)b[k0];
    f[2 * i + 1] = (h16)b[k0 + 1];
  }
  return f;
}

__device__ __forceinline__ void st_acc(float* __restrict__ p, int stride, v8f c) {
  int lane = threadIdx.x & 31;
  int col = lane & 15, half = lane >> 4;
#pragma unroll
  for (int i = 0; i < 8; ++i)
    p[(size_t)(i + 8 * half) * stride + col] = c[i];
}

__device__ __forceinline__ v8f ld_acc(const float* __restrict__ p, int stride) {
  int lane = threadIdx.x & 31;
  int col = lane & 15, half = lane >> 4;
  v8f c;
#pragma unroll
  for (int i = 0; i < 8; ++i)
    c[i] = p[(size_t)(i + 8 * half) * stride + col];
  return c;
}

// ---------------------------------------------------------------- GEMM
// C[M,N] = A[M,K] @ W[K,N], with W given pre-transposed+f16 as Bt[N,K].
// 256 threads = 8 waves in a 2x4 grid; each wave computes a 64x64 tile
// (4x4 WMMA accumulators = 128 VGPRs) -> 16 WMMAs per 16 b128 loads per K step.
// Epilogue: optional bias (len N), optional exact GELU, stores f32 and/or f16.
__device__ __forceinline__ void gemm_epi(v8f acc, int r0, int c0, int Ndim,
                                         const float* __restrict__ bias,
                                         float* __restrict__ Cf, h16* __restrict__ Ch,
                                         int act) {
  int lane = threadIdx.x & 31;
  int col = lane & 15, half = lane >> 4;
  int cc = c0 + col;
  float bval = bias ? bias[cc] : 0.f;
#pragma unroll
  for (int i = 0; i < 8; ++i) {
    float v = acc[i] + bval;
    if (act) v = 0.5f * v * (1.f + erff(v * 0.70710678118654752f));
    size_t off = (size_t)(r0 + i + 8 * half) * Ndim + cc;
    if (Cf) Cf[off] = v;
    if (Ch) Ch[off] = (h16)v;
  }
}

__global__ __launch_bounds__(256) void gemm_wmma(
    const h16* __restrict__ A, const h16* __restrict__ Bt,
    const float* __restrict__ bias, float* __restrict__ Cf, h16* __restrict__ Ch,
    int Mdim, int Ndim, int Kdim, int act) {
  int wave = threadIdx.x >> 5;
  int wm = wave >> 2;       // 0..1
  int wn = wave & 3;        // 0..3
  int m0 = blockIdx.y * 128 + wm * 64;
  int n0 = blockIdx.x * 256 + wn * 64;

  const h16* Ar[4];
  const h16* Br[4];
#pragma unroll
  for (int t = 0; t < 4; ++t) {
    Ar[t] = A + (size_t)(m0 + 16 * t) * Kdim;
    Br[t] = Bt + (size_t)(n0 + 16 * t) * Kdim;
  }

  v8f acc[4][4];
#pragma unroll
  for (int mi = 0; mi < 4; ++mi)
#pragma unroll
    for (int ni = 0; ni < 4; ++ni) acc[mi][ni] = zero8();

  for (int k = 0; k < Kdim; k += 32) {
    if (k + 128 < Kdim) {  // uniform; prefetch 256B down the K dim per fragment row
#pragma unroll
      for (int t = 0; t < 4; ++t) {
        pf_frag(Ar[t] + k, Kdim, 128);
        pf_frag(Br[t] + k, Kdim, 128);
      }
    }
    v16h fa[4], fb[4];
#pragma unroll
    for (int t = 0; t < 4; ++t) {
      fa[t] = ld_frag(Ar[t] + k, Kdim);
      fb[t] = ld_frag(Br[t] + k, Kdim);
    }
#pragma unroll
    for (int mi = 0; mi < 4; ++mi)
#pragma unroll
      for (int ni = 0; ni < 4; ++ni)
        acc[mi][ni] = wmma32(fa[mi], fb[ni], acc[mi][ni]);
  }

#pragma unroll
  for (int mi = 0; mi < 4; ++mi)
#pragma unroll
    for (int ni = 0; ni < 4; ++ni)
      gemm_epi(acc[mi][ni], m0 + 16 * mi, n0 + 16 * ni, Ndim, bias, Cf, Ch, act);
}

// ---------------------------------------------------------------- weight transpose + f32->f16
// Wt[n*K + k] = (h16) W[k*N + n]
__global__ __launch_bounds__(256) void tcvt_kernel(const float* __restrict__ W,
                                                   h16* __restrict__ Wt,
                                                   int Kdim, int Ndim) {
  __shared__ h16 tile[32][33];
  int nb = blockIdx.x * 32, kb = blockIdx.y * 32;
  int tx = threadIdx.x & 31, ty = threadIdx.x >> 5;  // 32 x 8
  for (int r = ty; r < 32; r += 8)
    tile[r][tx] = (h16)W[(size_t)(kb + r) * Ndim + nb + tx];
  __syncthreads();
  for (int r = ty; r < 32; r += 8)
    Wt[(size_t)(nb + r) * Kdim + kb + tx] = tile[tx][r];
}

// ---------------------------------------------------------------- fused attention
// One wave per (b, h, 16-query tile). LDS holds a 16 x klen f32 score row-block.
// Self-attn (do_rel=1): scores = (AC + rel_shift(BD)) * scale with causal+mem mask.
// Cross-attn (do_rel=0): plain scaled dot-product, no mask.
__global__ __launch_bounds__(32) void attn_kernel(
    const h16* __restrict__ qu, const h16* __restrict__ qv,
    const h16* __restrict__ kk, const h16* __restrict__ vv,
    const h16* __restrict__ rr, h16* __restrict__ out,
    int klenk, int do_rel, int maskM, float scale) {
  __shared__ float s[16 * kKLEN];  // 64 KB (max klen)
  const int lane = threadIdx.x;
  const int bh = blockIdx.y;
  const int b = bh / kH, h = bh % kH;
  const int i0 = blockIdx.x * 16;

  const h16* qub = qu + ((size_t)(b * kQ + i0) * kH + h) * kDH;
  const h16* kb  = kk + ((size_t)b * klenk * kH + h) * kDH;
  const h16* vb  = vv + ((size_t)b * klenk * kH + h) * kDH;

  if (do_rel) {
    // Pass 1: raw BD[i, jr] = (q_i + v_bias) . r_jr  for all jr in [0, klen)
    const h16* qvb = qv + ((size_t)(b * kQ + i0) * kH + h) * kDH;
    v16h aq0 = ld_frag(qvb, kD);
    v16h aq1 = ld_frag(qvb + 32, kD);
    const h16* rb = rr + h * kDH;
    for (int jt = 0; jt < klenk; jt += 16) {
      v8f acc = zero8();
      acc = wmma32(aq0, ld_frag(rb + (size_t)jt * kD, kD), acc);
      acc = wmma32(aq1, ld_frag(rb + (size_t)jt * kD + 32, kD), acc);
      st_acc(s + jt, klenk, acc);
    }
    __syncthreads();
    // TransformerXL rel-shift, exact: BD'[i,j] = BD[i, j + Q-1-i] (OOB -> 0, masked anyway).
    // One lane per row, ascending j: reads always at index >= write index -> safe in place.
    if (lane < 16) {
      int i = i0 + lane;
      int dsh = kQ - 1 - i;
      float* row = s + (size_t)lane * klenk;
      for (int j = 0; j < klenk; ++j) {
        int sj = j + dsh;
        row[j] = (sj < klenk) ? row[sj] : 0.f;
      }
    }
    __syncthreads();
  }
  {
    // Pass 2: accumulate AC on top (C operand sourced from LDS for self-attn).
    v16h aq0 = ld_frag(qub, kD);
    v16h aq1 = ld_frag(qub + 32, kD);
    for (int jt = 0; jt < klenk; jt += 16) {
      v8f acc = do_rel ? ld_acc(s + jt, klenk) : zero8();
      acc = wmma32(aq0, ld_frag(kb + (size_t)jt * kD, kD), acc);
      acc = wmma32(aq1, ld_frag(kb + (size_t)jt * kD + 32, kD), acc);
      st_acc(s + jt, klenk, acc);
    }
  }
  __syncthreads();
  // Softmax: 2 lanes per row (halves of klen), combine via shfl_xor(16).
  {
    int row = lane & 15, half = lane >> 4;
    int i = i0 + row;
    int hlen = klenk >> 1;
    int j0 = half * hlen;
    float* srow = s + (size_t)row * klenk;
    float mx = -1e30f;
    for (int j = j0; j < j0 + hlen; ++j) {
      float v = srow[j] * scale;
      if (do_rel && j > i + maskM) v = -1e30f;
      srow[j] = v;
      mx = fmaxf(mx, v);
    }
    mx = fmaxf(mx, __shfl_xor(mx, 16, 32));
    float sum = 0.f;
    for (int j = j0; j < j0 + hlen; ++j) {
      float e = __expf(srow[j] - mx);   // max-subtracted -> fast path is safe
      srow[j] = e;
      sum += e;
    }
    sum += __shfl_xor(sum, 16, 32);
    float rinv = 1.f / sum;
    for (int j = j0; j < j0 + hlen; ++j) srow[j] *= rinv;
  }
  __syncthreads();
  // P @ V : A = P (f32 in LDS -> f16), B = V with strided K (keys).
  h16* ob = out + ((size_t)(b * kQ + i0) * kH + h) * kDH;
  for (int dt = 0; dt < 4; ++dt) {
    v8f acc = zero8();
    for (int kc = 0; kc < klenk; kc += 32) {
      v16h pa = ld_fragA_f32(s + kc, klenk);
      v16h pb = ld_fragB_strided(vb + (size_t)kc * kD + dt * 16, kD);
      acc = wmma32(pa, pb, acc);
    }
    int col = lane & 15, half = lane >> 4;
#pragma unroll
    for (int i = 0; i < 8; ++i)
      ob[(size_t)(i + 8 * half) * kD + dt * 16 + col] = (h16)acc[i];
  }
}

// ---------------------------------------------------------------- elementwise kernels
__global__ void embed_kernel(const float* __restrict__ emb, const int* __restrict__ ids,
                             float* __restrict__ x, h16* __restrict__ x16) {
  int row = blockIdx.x;
  int id = ids[row];
  for (int d = threadIdx.x; d < kD; d += blockDim.x) {
    float v = emb[(size_t)id * kD + d] * 32.0f;  // sqrt(D) = 32
    x[(size_t)row * kD + d] = v;
    x16[(size_t)row * kD + d] = (h16)v;
  }
}

__global__ void posenc_kernel(h16* __restrict__ r16) {
  int jr = blockIdx.x;
  float pos = (float)(kKLEN - 1 - jr);
  for (int d = threadIdx.x; d < kD; d += blockDim.x) {
    int i = (d < kD / 2) ? d : d - kD / 2;
    float freq = expf(-logf(10000.f) * ((float)(2 * i) / (float)kD));
    float val = pos * freq;
    r16[(size_t)jr * kD + d] = (h16)((d < kD / 2) ? sinf(val) : cosf(val));
  }
}

__global__ void cvt_kernel(const float* __restrict__ src, h16* __restrict__ dst, size_t n) {
  size_t i = (size_t)blockIdx.x * blockDim.x + threadIdx.x;
  if (i < n) dst[i] = (h16)src[i];
}

__global__ void adduv_kernel(const float* __restrict__ q, const float* __restrict__ u,
                             const float* __restrict__ vb,
                             h16* __restrict__ qu16, h16* __restrict__ qv16, size_t n) {
  size_t i = (size_t)blockIdx.x * blockDim.x + threadIdx.x;
  if (i < n) {
    int d = (int)(i % kD);
    float qq = q[i];
    qu16[i] = (h16)(qq + u[d]);
    qv16[i] = (h16)(qq + vb[d]);
  }
}

__global__ void concat_kernel(const float* __restrict__ mems_l, const float* __restrict__ x,
                              h16* __restrict__ mem16) {
  int row = blockIdx.x;              // 0 .. B*KLEN-1
  int b = row / kKLEN, j = row % kKLEN;
  const float* src = (j < kM) ? (mems_l + ((size_t)b * kM + j) * kD)
                              : (x + ((size_t)b * kQ + (j - kM)) * kD);
  h16* dst = mem16 + (size_t)row * kD;
  for (int d = threadIdx.x; d < kD; d += blockDim.x) dst[d] = (h16)src[d];
}

// x = LN(x + y + bias) * g + b ; also emits f16 copy.
__global__ __launch_bounds__(256) void ln_fuse(
    float* __restrict__ x, const float* __restrict__ y, const float* __restrict__ bias,
    const float* __restrict__ g, const float* __restrict__ bb, h16* __restrict__ x16) {
  __shared__ float sA[256], sB[256];
  int row = blockIdx.x, tid = threadIdx.x;
  float ps = 0.f, pq = 0.f;
  for (int d = tid; d < kD; d += 256) {
    float t = x[(size_t)row * kD + d] + y[(size_t)row * kD + d] + bias[d];
    ps += t;
    pq += t * t;
  }
  sA[tid] = ps; sB[tid] = pq;
  __syncthreads();
  for (int s2 = 128; s2 > 0; s2 >>= 1) {
    if (tid < s2) { sA[tid] += sA[tid + s2]; sB[tid] += sB[tid + s2]; }
    __syncthreads();
  }
  float mean = sA[0] / kD;
  float var = sB[0] / kD - mean * mean;
  float rstd = rsqrtf(var + 1e-5f);
  for (int d = tid; d < kD; d += 256) {
    float t = x[(size_t)row * kD + d] + y[(size_t)row * kD + d] + bias[d];
    float o = (t - mean) * rstd * g[d] + bb[d];
    x[(size_t)row * kD + d] = o;
    x16[(size_t)row * kD + d] = (h16)o;
  }
}

// ---------------------------------------------------------------- launch
extern "C" void kernel_launch(void* const* d_in, const int* in_sizes, int n_in,
                              void* d_out, int out_size, void* d_ws, size_t ws_size,
                              hipStream_t stream) {
  const float* enc    = (const float*)d_in[0];
  const float* mems   = (const float*)d_in[1];
  const float* emb    = (const float*)d_in[2];
  const float* u      = (const float*)d_in[3];
  const float* vbias  = (const float*)d_in[4];
  const float* sa_Wq  = (const float*)d_in[5];
  const float* sa_Wk  = (const float*)d_in[6];
  const float* sa_Wv  = (const float*)d_in[7];
  const float* sa_Wr  = (const float*)d_in[8];
  const float* sa_fcW = (const float*)d_in[9];
  const float* sa_fcB = (const float*)d_in[10];
  const float* ln1_g  = (const float*)d_in[11];
  const float* ln1_b  = (const float*)d_in[12];
  const float* ca_Wq  = (const float*)d_in[13];
  const float* ca_Wk  = (const float*)d_in[14];
  const float* ca_Wv  = (const float*)d_in[15];
  const float* ca_fcW = (const float*)d_in[16];
  const float* ca_fcB = (const float*)d_in[17];
  const float* ln2_g  = (const float*)d_in[18];
  const float* ln2_b  = (const float*)d_in[19];
  const float* ff_W1  = (const float*)d_in[20];
  const float* ff_b1  = (const float*)d_in[21];
  const float* ff_W2  = (const float*)d_in[22];
  const float* ff_b2  = (const float*)d_in[23];
  const float* ln3_g  = (const float*)d_in[24];
  const float* ln3_b  = (const float*)d_in[25];
  const float* out_W  = (const float*)d_in[26];
  const int*   ids    = (const int*)d_in[27];

  // bump allocator over workspace (~105 MB used)
  char* p = (char*)d_ws;
  auto alloc = [&](size_t bytes) -> void* {
    void* r = (void*)p;
    p += (bytes + 255) & ~(size_t)255;
    return r;
  };
  float* xf     = (float*)alloc((size_t)kBQ * kD * 4);
  h16*   x16    = (h16*)  alloc((size_t)kBQ * kD * 2);
  h16*   r16    = (h16*)  alloc((size_t)kKLEN * kD * 2);
  h16*   enc16  = (h16*)  alloc((size_t)kB * kE * kD * 2);
  h16*   mem16  = (h16*)  alloc((size_t)kB * kKLEN * kD * 2);
  float* qf     = (float*)alloc((size_t)kBQ * kD * 4);
  h16*   qu16   = (h16*)  alloc((size_t)kBQ * kD * 2);
  h16*   qv16   = (h16*)  alloc((size_t)kBQ * kD * 2);
  h16*   k16    = (h16*)  alloc((size_t)kB * kKLEN * kD * 2);
  h16*   v16    = (h16*)  alloc((size_t)kB * kKLEN * kD * 2);
  h16*   r16p   = (h16*)  alloc((size_t)kKLEN * kD * 2);
  h16*   av16   = (h16*)  alloc((size_t)kBQ * kD * 2);
  float* gout   = (float*)alloc((size_t)kBQ * kD * 4);
  h16*   h16buf = (h16*)  alloc((size_t)kBQ * kFF * 2);
  h16*   Wt     = (h16*)  alloc((size_t)kV * kD * 2);   // largest transposed weight

  auto gemm = [&](const h16* A, const h16* Bt, const float* bias, float* Cf, h16* Ch,
                  int M_, int N_, int K_, int act) {
    dim3 g(N_ / 256, M_ / 128);
    gemm_wmma<<<g, 256, 0, stream>>>(A, Bt, bias, Cf, Ch, M_, N_, K_, act);
  };
  auto tcvt = [&](const float* W, int K_, int N_) {
    dim3 g(N_ / 32, K_ / 32);
    tcvt_kernel<<<g, 256, 0, stream>>>(W, Wt, K_, N_);
  };

  const float scale = 0.125f;  // DH^-0.5
  const dim3 attn_grid(kQ / 16, kB * kH);

  // ---- prologue
  embed_kernel<<<kBQ, 256, 0, stream>>>(emb, ids, xf, x16);
  posenc_kernel<<<kKLEN, 256, 0, stream>>>(r16);
  {
    size_t n = (size_t)kB * kE * kD;
    cvt_kernel<<<(unsigned)((n + 255) / 256), 256, 0, stream>>>(enc, enc16, n);
  }

  for (int l = 0; l < kL; ++l) {
    const size_t DD = (size_t)kD * kD;
    // ---- self-attention (TransformerXL relative)
    concat_kernel<<<kB * kKLEN, 256, 0, stream>>>(mems + (size_t)l * kB * kM * kD, xf, mem16);
    tcvt(sa_Wq + l * DD, kD, kD);
    gemm(x16, Wt, nullptr, qf, nullptr, kBQ, kD, kD, 0);
    {
      size_t n = (size_t)kBQ * kD;
      adduv_kernel<<<(unsigned)((n + 255) / 256), 256, 0, stream>>>(qf, u, vbias, qu16, qv16, n);
    }
    tcvt(sa_Wk + l * DD, kD, kD);
    gemm(mem16, Wt, nullptr, nullptr, k16, kB * kKLEN, kD, kD, 0);
    tcvt(sa_Wv + l * DD, kD, kD);
    gemm(mem16, Wt, nullptr, nullptr, v16, kB * kKLEN, kD, kD, 0);
    tcvt(sa_Wr + l * DD, kD, kD);
    gemm(r16, Wt, nullptr, nullptr, r16p, kKLEN, kD, kD, 0);
    attn_kernel<<<attn_grid, 32, 0, stream>>>(qu16, qv16, k16, v16, r16p, av16,
                                              kKLEN, 1, kM, scale);
    tcvt(sa_fcW + l * DD, kD, kD);
    gemm(av16, Wt, nullptr, gout, nullptr, kBQ, kD, kD, 0);
    ln_fuse<<<kBQ, 256, 0, stream>>>(xf, gout, sa_fcB + l * kD, ln1_g + l * kD,
                                     ln1_b + l * kD, x16);

    // ---- cross-attention (standard)
    tcvt(ca_Wq + l * DD, kD, kD);
    gemm(x16, Wt, nullptr, nullptr, qu16, kBQ, kD, kD, 0);
    tcvt(ca_Wk + l * DD, kD, kD);
    gemm(enc16, Wt, nullptr, nullptr, k16, kB * kE, kD, kD, 0);
    tcvt(ca_Wv + l * DD, kD, kD);
    gemm(enc16, Wt, nullptr, nullptr, v16, kB * kE, kD, kD, 0);
    attn_kernel<<<attn_grid, 32, 0, stream>>>(qu16, nullptr, k16, v16, nullptr, av16,
                                              kE, 0, 0, scale);
    tcvt(ca_fcW + l * DD, kD, kD);
    gemm(av16, Wt, nullptr, gout, nullptr, kBQ, kD, kD, 0);
    ln_fuse<<<kBQ, 256, 0, stream>>>(xf, gout, ca_fcB + l * kD, ln2_g + l * kD,
                                     ln2_b + l * kD, x16);

    // ---- feed-forward: GELU(x@W1+b1)@W2 + residual LN
    tcvt(ff_W1 + (size_t)l * kD * kFF, kD, kFF);
    gemm(x16, Wt, ff_b1 + (size_t)l * kFF, nullptr, h16buf, kBQ, kFF, kD, 1);
    tcvt(ff_W2 + (size_t)l * kFF * kD, kFF, kD);
    gemm(h16buf, Wt, nullptr, gout, nullptr, kBQ, kD, kFF, 0);
    ln_fuse<<<kBQ, 256, 0, stream>>>(xf, gout, ff_b2 + l * kD, ln3_g + l * kD,
                                     ln3_b + l * kD, x16);
  }

  // ---- output head: [2048,1024] @ [1024,8192] -> d_out (f32)
  tcvt(out_W, kD, kV);
  gemm(x16, Wt, nullptr, (float*)d_out, nullptr, kBQ, kV, kD, 0);
}
